// MEGNet_893353198391
// MI455X (gfx1250) — compile-verified, matrix operand order checked by reading, not run on hardware
//
#include <hip/hip_runtime.h>
#include <cstddef>

// ---------------------------------------------------------------------------
// MEGNet forward for MI455X (gfx1250, wave32, WMMA bf16)
// ---------------------------------------------------------------------------

typedef __attribute__((ext_vector_type(16))) __bf16 v16bf;
typedef __attribute__((ext_vector_type(8)))  __bf16 v8bf;
typedef __attribute__((ext_vector_type(8)))  float  v8f;

#define N_NODES 20000
#define N_EDGES 320000
#define EDGE_CH 65536   // edge-row chunk (multiple of 16; 320000 = 4*65536 + 57856)

__device__ __forceinline__ float silu_f(float x) { return x / (1.0f + __expf(-x)); }

// ---------------------------------------------------------------------------
// Core GEMM: C[M,N] = act( A[M,KP] * W + bias ), W packed as Wt (N x KP) bf16
// row-major (transposed, zero-padded K).  KP compile-time in {128,256}.
// Block = 128 threads = 4 waves; each wave computes a 16x32 tile (two 16x16
// WMMA subtiles sharing one A fragment); block tile = 16 x 128.
// A fragment (16-bit A 16x32): lane m = lane&15, kb = (lane>>4)*8;
// elems [0..7] = K kb+0..7, [8..15] = K kb+16..23.  B loaded identically
// from Wt rows (W transposed).  All loads are 16-byte vector loads.
// ---------------------------------------------------------------------------
template <int KP>
__global__ void gemm_bf16_wmma(const float* __restrict__ A, int lda,
                               const __bf16* __restrict__ Wt,
                               const float* __restrict__ bias,
                               float* __restrict__ C, int N, int act)
{
    const int wave = threadIdx.x >> 5;
    const int lane = threadIdx.x & 31;
    const int row0 = blockIdx.x * 16;
    const int col0 = blockIdx.y * 128 + wave * 32;     // two 16-wide subtiles
    const int m  = lane & 15;
    const int kb = (lane >> 4) * 8;

    const float*  arow  = A  + (size_t)(row0 + m) * lda + kb;
    const __bf16* wrow0 = Wt + (size_t)(col0 + m) * KP + kb;
    const __bf16* wrow1 = wrow0 + (size_t)16 * KP;

    v8f acc0 = {};
    v8f acc1 = {};
#pragma unroll
    for (int k0 = 0; k0 < KP; k0 += 32) {
        const float4 a0 = *(const float4*)(arow + k0);
        const float4 a1 = *(const float4*)(arow + k0 + 4);
        const float4 a2 = *(const float4*)(arow + k0 + 16);
        const float4 a3 = *(const float4*)(arow + k0 + 20);
        v16bf a;
        a[0]  = (__bf16)a0.x; a[1]  = (__bf16)a0.y; a[2]  = (__bf16)a0.z; a[3]  = (__bf16)a0.w;
        a[4]  = (__bf16)a1.x; a[5]  = (__bf16)a1.y; a[6]  = (__bf16)a1.z; a[7]  = (__bf16)a1.w;
        a[8]  = (__bf16)a2.x; a[9]  = (__bf16)a2.y; a[10] = (__bf16)a2.z; a[11] = (__bf16)a2.w;
        a[12] = (__bf16)a3.x; a[13] = (__bf16)a3.y; a[14] = (__bf16)a3.z; a[15] = (__bf16)a3.w;

        const v8bf b0lo = *(const v8bf*)(wrow0 + k0);
        const v8bf b0hi = *(const v8bf*)(wrow0 + k0 + 16);
        const v8bf b1lo = *(const v8bf*)(wrow1 + k0);
        const v8bf b1hi = *(const v8bf*)(wrow1 + k0 + 16);
        v16bf b0, b1;
#pragma unroll
        for (int j = 0; j < 8; ++j) {
            b0[j] = b0lo[j]; b0[8 + j] = b0hi[j];
            b1[j] = b1lo[j]; b1[8 + j] = b1hi[j];
        }
        acc0 = __builtin_amdgcn_wmma_f32_16x16x32_bf16(false, a, false, b0,
                                                       (short)0, acc0, false, false);
        acc1 = __builtin_amdgcn_wmma_f32_16x16x32_bf16(false, a, false, b1,
                                                       (short)0, acc1, false, false);
    }

    const int colw0 = col0 + m;
    const int rbase = row0 + ((lane >> 4) * 8);
    const float bv0 = bias ? bias[colw0] : 0.0f;
    const float bv1 = bias ? bias[colw0 + 16] : 0.0f;
#pragma unroll
    for (int r = 0; r < 8; ++r) {
        float v0 = acc0[r] + bv0;
        float v1 = acc1[r] + bv1;
        if (act) { v0 = silu_f(v0); v1 = silu_f(v1); }
        float* crow = C + (size_t)(rbase + r) * N;
        crow[colw0]      = v0;
        crow[colw0 + 16] = v1;
    }
}

// Pack fp32 W rows [rowOff, rowOff+Kact) of a (Ktot x N) row-major matrix into
// bf16 Wt (N x Kp) row-major, zero-padding K up to Kp.
__global__ void pack_weight(const float* __restrict__ W, int N, int rowOff,
                            int Kact, int Kp, __bf16* __restrict__ Wt)
{
    size_t i = (size_t)blockIdx.x * blockDim.x + threadIdx.x;
    if (i >= (size_t)N * Kp) return;
    int n = (int)(i / Kp), k = (int)(i % Kp);
    float v = (k < Kact) ? W[(size_t)(rowOff + k) * N + n] : 0.0f;
    Wt[i] = (__bf16)v;
}

// Zero-pad rows: Y (rows x Kp) <- X (rows x K), tail zero
__global__ void pad_rows(const float* __restrict__ X, int K, int Kp,
                         float* __restrict__ Y, size_t rows)
{
    size_t i = (size_t)blockIdx.x * blockDim.x + threadIdx.x;
    if (i >= rows * (size_t)Kp) return;
    size_t r = i / Kp;
    int k = (int)(i % Kp);
    Y[i] = (k < K) ? X[r * K + k] : 0.0f;
}

// ------------------------------ utility kernels ----------------------------
__global__ void fill_kernel(float* p, size_t n, float v)
{
    size_t i = (size_t)blockIdx.x * blockDim.x + threadIdx.x;
    if (i < n) p[i] = v;
}
__global__ void axpy_kernel(float* a, const float* b, size_t n)
{
    size_t i = (size_t)blockIdx.x * blockDim.x + threadIdx.x;
    if (i < n) a[i] += b[i];
}
__global__ void combine_silu(float* a, const float* b, size_t n)
{
    size_t i = (size_t)blockIdx.x * blockDim.x + threadIdx.x;
    if (i < n) a[i] = silu_f(a[i] + b[i]);
}
__global__ void deg_kernel(const int* __restrict__ dst, int ne, float* deg)
{
    int e = blockIdx.x * blockDim.x + threadIdx.x;
    if (e < ne) atomicAdd(&deg[dst[e]], 1.0f);
}
__global__ void ve_div(float* ve, const float* __restrict__ deg, int nn)
{
    size_t i = (size_t)blockIdx.x * blockDim.x + threadIdx.x;
    if (i < (size_t)nn * 128) ve[i] /= fmaxf(deg[i >> 7], 1.0f);
}
// T1[e,:256] = silu(T1 + NS1[src[e]] + NS2[dst[e]])   (decomposed concat layer)
__global__ void edge_gather_silu(float* __restrict__ T1,
                                 const float* __restrict__ NS1,
                                 const float* __restrict__ NS2,
                                 const int* __restrict__ src,
                                 const int* __restrict__ dst, int e0, int ne)
{
    size_t t = (size_t)blockIdx.x * blockDim.x + threadIdx.x;
    int e = (int)(t >> 6);            // 64 threads per edge, 4 features each
    if (e >= ne) return;
    int f = (int)(t & 63) * 4;
    int s = src[e0 + e], d = dst[e0 + e];
    float* tr = T1 + (size_t)e * 256 + f;
    const float* r1 = NS1 + (size_t)s * 256 + f;
    const float* r2 = NS2 + (size_t)d * 256 + f;
#pragma unroll
    for (int i = 0; i < 4; ++i) tr[i] = silu_f(tr[i] + r1[i] + r2[i]);
}
// segment-sum ef2 rows by dst
__global__ void scatter_edges(const float* __restrict__ EF2c,
                              const int* __restrict__ dst, int e0, int ne,
                              float* __restrict__ nsum)
{
    size_t t = (size_t)blockIdx.x * blockDim.x + threadIdx.x;
    int e = (int)(t >> 5);            // 32 threads per edge, 4 features each
    if (e >= ne) return;
    int f = (int)(t & 31) * 4;
    int d = dst[e0 + e];
    const float* row = EF2c + (size_t)e * 128 + f;
#pragma unroll
    for (int i = 0; i < 4; ++i) atomicAdd(&nsum[(size_t)d * 128 + f + i], row[i]);
}
// column sums of an (M x 128) matrix into out[128] (pre-zeroed)
__global__ void colsum_kernel(const float* __restrict__ X, int M, float* out)
{
    __shared__ float lds[256];
    int t = threadIdx.x, col = t & 127, half = t >> 7;
    size_t r0 = (size_t)blockIdx.x * 512;
    float s = 0.0f;
    for (size_t r = r0 + half; r < r0 + 512 && r < (size_t)M; r += 2)
        s += X[r * 128 + col];
    lds[t] = s;
    __syncthreads();
    if (t < 128) atomicAdd(&out[col], lds[t] + lds[t + 128]);
}
// M=1 dense layer: y[n] = act(b[n] + sum_k x[k] * W[k*N+n]),  W is (K x N)
__global__ void vec_mlp_layer(const float* __restrict__ x, const float* __restrict__ W,
                              const float* __restrict__ b, float* __restrict__ y,
                              int K, int N, int act)
{
    int n = blockIdx.x * blockDim.x + threadIdx.x;
    if (n >= N) return;
    float s = b ? b[n] : 0.0f;
    for (int k = 0; k < K; ++k) s += x[k] * W[(size_t)k * N + n];
    y[n] = act ? silu_f(s) : s;
}
__global__ void assemble_avec(const float* esum, const float* nsum, const float* u,
                              float* avec, float invE, float invN)
{
    int t = threadIdx.x;               // 384 threads
    if (t < 128)       avec[t] = esum[t] * invE;
    else if (t < 256)  avec[t] = nsum[t - 128] * invN;
    else if (t < 384)  avec[t] = u[t - 256];
}
// torch-order LSTM step (i,f,g,o).  Single block of 512 threads.
__global__ void lstm_step(const float* __restrict__ q, const float* __restrict__ Wih,
                          const float* __restrict__ Whh, const float* __restrict__ bih,
                          const float* __restrict__ bhh, float* h, float* c)
{
    __shared__ float z[512];
    int j = threadIdx.x;
    float s = bih[j] + bhh[j];
    for (int k = 0; k < 256; ++k) s += q[k] * Wih[(size_t)j * 256 + k];
    for (int k = 0; k < 128; ++k) s += h[k] * Whh[(size_t)j * 128 + k];
    z[j] = s;
    __syncthreads();
    if (j < 128) {
        float ig = 1.0f / (1.0f + __expf(-z[j]));
        float fg = 1.0f / (1.0f + __expf(-z[128 + j]));
        float gg = tanhf(z[256 + j]);
        float og = 1.0f / (1.0f + __expf(-z[384 + j]));
        float cn = fg * c[j] + ig * gg;
        c[j] = cn;
        h[j] = og * tanhf(cn);
    }
}
__device__ void atomicMaxFloat(float* addr, float val)
{
    unsigned int* ai = (unsigned int*)addr;
    unsigned int old = *ai;
    while (true) {
        float f = __uint_as_float(old);
        if (f >= val) break;
        unsigned int assumed = old;
        old = atomicCAS(ai, assumed, __float_as_uint(val));
        if (old == assumed) break;
    }
}
__global__ void s2s_logits(const float* __restrict__ feat, int n,
                           const float* __restrict__ h, float* e, float* gmax)
{
    __shared__ float hl[128];
    __shared__ float red[256];
    int t = threadIdx.x;
    if (t < 128) hl[t] = h[t];
    __syncthreads();
    int r = blockIdx.x * 256 + t;
    float s = -3.0e38f;
    if (r < n) {
        s = 0.0f;
        const float* row = feat + (size_t)r * 128;
        for (int f = 0; f < 128; ++f) s += row[f] * hl[f];
        e[r] = s;
    }
    red[t] = s;
    __syncthreads();
    for (int w = 128; w > 0; w >>= 1) {
        if (t < w) red[t] = fmaxf(red[t], red[t + w]);
        __syncthreads();
    }
    if (t == 0) atomicMaxFloat(gmax, red[0]);
}
__global__ void s2s_accum(const float* __restrict__ feat, const float* __restrict__ e,
                          int n, const float* __restrict__ gmax,
                          float* racc, float* psum)
{
    __shared__ float p[256];
    int t = threadIdx.x;
    int r = blockIdx.x * 256 + t;
    float mv = *gmax;
    p[t] = (r < n) ? __expf(e[r] - mv) : 0.0f;
    __syncthreads();
    int rows = n - blockIdx.x * 256;
    if (rows > 256) rows = 256;
    if (t < 128) {
        float acc = 0.0f;
        size_t base = (size_t)blockIdx.x * 256 * 128;
        for (int i = 0; i < rows; ++i) acc += p[i] * feat[base + (size_t)i * 128 + t];
        atomicAdd(&racc[t], acc);
    } else if (t == 128) {
        float s = 0.0f;
        for (int i = 0; i < rows; ++i) s += p[i];
        atomicAdd(psum, s);
    }
}
__global__ void qstar_update(const float* h, const float* racc, const float* psum, float* q)
{
    int t = threadIdx.x;
    if (t < 128) q[t] = h[t];
    else         q[t] = racc[t - 128] / psum[0];
}
// final head: [nq(256), eq(256), u(128)] -> 128 silu -> 64 silu -> 1
__global__ void out_mlp(const float* nq, const float* eq, const float* u,
                        const float* W0, const float* b0, const float* W1,
                        const float* b1, const float* W2, const float* b2, float* out)
{
    __shared__ float x[640];
    __shared__ float h1[128];
    __shared__ float h2[64];
    int t = threadIdx.x;                  // 128 threads
    for (int i = t; i < 640; i += 128)
        x[i] = (i < 256) ? nq[i] : (i < 512) ? eq[i - 256] : u[i - 512];
    __syncthreads();
    {
        float s = b0[t];
        for (int k = 0; k < 640; ++k) s += x[k] * W0[(size_t)k * 128 + t];
        h1[t] = silu_f(s);
    }
    __syncthreads();
    if (t < 64) {
        float s = b1[t];
        for (int k = 0; k < 128; ++k) s += h1[k] * W1[(size_t)k * 64 + t];
        h2[t] = silu_f(s);
    }
    __syncthreads();
    if (t == 0) {
        float s = b2[0];
        for (int k = 0; k < 64; ++k) s += h2[k] * W2[k];
        out[0] = s;
    }
}

// ---------------------------------------------------------------------------
extern "C" void kernel_launch(void* const* d_in, const int* in_sizes, int n_in,
                              void* d_out, int out_size, void* d_ws, size_t ws_size,
                              hipStream_t stream)
{
    (void)in_sizes; (void)n_in; (void)out_size; (void)ws_size;

    const float* edge_feat  = (const float*)d_in[0];   // 320000 x 100
    const float* node_feat  = (const float*)d_in[1];   // 20000 x 128
    const float* graph_attr = (const float*)d_in[2];   // 1 x 32
    const int*   src        = (const int*)d_in[3];
    const int*   dst        = (const int*)d_in[4];
    auto P = [&](int i) { return (const float*)d_in[i]; };

    // Param leaf indices (jax tree order: sorted dict keys, list order).
    const int ATTR_ENC = 5;                       // W0,b0,W1,b1
    const int convA[3] = { 9, 27, 57 };           // conv attr  mlp (6 leaves)
    const int convE[3] = { 15, 33, 63 };          // conv edge  mlp (6 leaves)
    const int convN[3] = { 21, 39, 69 };          // conv node  mlp (6 leaves)
    const int preA[3]  = { -1, 45, 75 };          // pre attr (4 leaves)
    const int preE[3]  = { -1, 49, 79 };
    const int preN[3]  = { -1, 53, 83 };
    const int EDGE_ENC = 87, EDGE_S2S = 91, NODE_ENC = 95, NODE_S2S = 99, OUT_MLP = 103;

    // ---------------- workspace layout (f32 region, then bf16 weights) -----
    float* wbase = (float*)d_ws;
    size_t off = 0;
    auto alloc = [&](size_t n) { float* p = wbase + off; off += n; return p; };
    float* EF   = alloc((size_t)N_EDGES * 128);
    float* EF2  = alloc((size_t)N_EDGES * 128);
    float* EFPb = alloc((size_t)N_EDGES * 128);   // pre-MLP out; also padded enc input
    float* NF   = alloc((size_t)N_NODES * 128);
    float* NF2  = alloc((size_t)N_NODES * 128);
    float* NFPb = alloc((size_t)N_NODES * 128);
    float* NS1  = alloc((size_t)N_NODES * 256);
    float* NS2  = alloc((size_t)N_NODES * 256);
    float* NT1  = alloc((size_t)N_NODES * 256);
    float* NT2  = alloc((size_t)N_NODES * 256);
    float* VE   = alloc((size_t)N_NODES * 128);
    float* T1   = alloc((size_t)EDGE_CH * 256);
    float* T2   = alloc((size_t)EDGE_CH * 256);
    float* LOGI = alloc(N_EDGES);
    float* DEG  = alloc(N_NODES);
    float* U    = alloc(128);
    float* U2   = alloc(128);
    float* UPb  = alloc(128);
    float* UB_E = alloc(256);
    float* UB_N = alloc(256);
    float* ESUM = alloc(128);
    float* NSUM = alloc(128);
    float* AVEC = alloc(384);
    float* AT1  = alloc(256);
    float* AT2  = alloc(256);
    float* Hs   = alloc(128);
    float* Cs   = alloc(128);
    float* RACC = alloc(128);
    float* NODEQ = alloc(256);
    float* EDGEQ = alloc(256);
    float* PSUM = alloc(1);
    float* GMAX = alloc(1);

    size_t fbytes = (off * sizeof(float) + 255) & ~(size_t)255;
    __bf16* wp = (__bf16*)((char*)d_ws + fbytes);
    size_t woff = 0;

    auto packW = [&](int idx, int N, int rowOff, int Kact, int Kp) -> const __bf16* {
        __bf16* dstp = wp + woff;
        woff += (size_t)N * Kp;
        size_t tot = (size_t)N * Kp;
        pack_weight<<<(unsigned)((tot + 255) / 256), 256, 0, stream>>>(P(idx), N, rowOff, Kact, Kp, dstp);
        return dstp;
    };
    auto gemm = [&](const float* A, int lda, int Kp, const __bf16* Wt,
                    const float* bias, float* C, int M, int N, int act) {
        dim3 g((unsigned)(M / 16), (unsigned)(N / 128));
        if (Kp == 128)
            gemm_bf16_wmma<128><<<g, dim3(128), 0, stream>>>(A, lda, Wt, bias, C, N, act);
        else
            gemm_bf16_wmma<256><<<g, dim3(128), 0, stream>>>(A, lda, Wt, bias, C, N, act);
    };
    auto fill = [&](float* p, size_t n, float v) {
        fill_kernel<<<(unsigned)((n + 255) / 256), 256, 0, stream>>>(p, n, v);
    };
    auto vml = [&](const float* x, const float* W, const float* b, float* y, int K, int N, int act) {
        vec_mlp_layer<<<(unsigned)((N + 255) / 256), 256, 0, stream>>>(x, W, b, y, K, N, act);
    };
    auto axpy = [&](float* a, const float* b, size_t n) {
        axpy_kernel<<<(unsigned)((n + 255) / 256), 256, 0, stream>>>(a, b, n);
    };

    // ---------------- pack all GEMM weights to bf16 (N x Kp) --------------
    const __bf16* ENC_E0 = packW(EDGE_ENC,     256, 0, 100, 128);
    const __bf16* ENC_E1 = packW(EDGE_ENC + 2, 128, 0, 256, 256);
    const __bf16* ENC_N0 = packW(NODE_ENC,     256, 0, 128, 128);
    const __bf16* ENC_N1 = packW(NODE_ENC + 2, 128, 0, 256, 256);
    struct BlkW {
        const __bf16 *esrc, *edst, *eef, *e2, *e3, *nve, *nnf, *n2, *n3;
        const __bf16 *pe0, *pe1, *pn0, *pn1;
    } bw[3];
    for (int b = 0; b < 3; ++b) {
        bw[b].esrc = packW(convE[b],     256, 0,   128, 128);   // rows   0..127 of 512x256
        bw[b].edst = packW(convE[b],     256, 128, 128, 128);   // rows 128..255
        bw[b].eef  = packW(convE[b],     256, 256, 128, 128);   // rows 256..383
        bw[b].e2   = packW(convE[b] + 2, 256, 0,   256, 256);
        bw[b].e3   = packW(convE[b] + 4, 128, 0,   256, 256);
        bw[b].nve  = packW(convN[b],     256, 0,   128, 128);   // rows   0..127 of 384x256
        bw[b].nnf  = packW(convN[b],     256, 128, 128, 128);   // rows 128..255
        bw[b].n2   = packW(convN[b] + 2, 256, 0,   256, 256);
        bw[b].n3   = packW(convN[b] + 4, 128, 0,   256, 256);
        if (b) {
            bw[b].pe0 = packW(preE[b],     256, 0, 128, 128);
            bw[b].pe1 = packW(preE[b] + 2, 128, 0, 256, 256);
            bw[b].pn0 = packW(preN[b],     256, 0, 128, 128);
            bw[b].pn1 = packW(preN[b] + 2, 128, 0, 256, 256);
        }
    }

    // ---------------- encoders --------------------------------------------
    vml(graph_attr, P(ATTR_ENC), P(ATTR_ENC + 1), AT1, 32, 256, 1);
    vml(AT1, P(ATTR_ENC + 2), P(ATTR_ENC + 3), U, 256, 128, 1);

    gemm(node_feat, 128, 128, ENC_N0, P(NODE_ENC + 1), NT1, N_NODES, 256, 1);
    gemm(NT1, 256, 256, ENC_N1, P(NODE_ENC + 3), NF, N_NODES, 128, 1);

    // zero-pad edge features (K=100 -> 128) into EFPb (free until block 1)
    pad_rows<<<(unsigned)(((size_t)N_EDGES * 128 + 255) / 256), 256, 0, stream>>>(
        edge_feat, 100, 128, EFPb, (size_t)N_EDGES);
    for (int r0 = 0; r0 < N_EDGES; r0 += EDGE_CH) {
        int ne = N_EDGES - r0; if (ne > EDGE_CH) ne = EDGE_CH;
        gemm(EFPb + (size_t)r0 * 128, 128, 128, ENC_E0, P(EDGE_ENC + 1), T1, ne, 256, 1);
        gemm(T1, 256, 256, ENC_E1, P(EDGE_ENC + 3), EF + (size_t)r0 * 128, ne, 128, 1);
    }

    fill(DEG, N_NODES, 0.0f);
    deg_kernel<<<(N_EDGES + 255) / 256, 256, 0, stream>>>(dst, N_EDGES, DEG);

    // ---------------- graph network blocks --------------------------------
    for (int b = 0; b < 3; ++b) {
        const float* EFp = EF;
        const float* NFp = NF;
        const float* Uc  = U;
        if (b) {
            gemm(NF, 128, 128, bw[b].pn0, P(preN[b] + 1), NT1, N_NODES, 256, 1);
            gemm(NT1, 256, 256, bw[b].pn1, P(preN[b] + 3), NFPb, N_NODES, 128, 1);
            for (int r0 = 0; r0 < N_EDGES; r0 += EDGE_CH) {
                int ne = N_EDGES - r0; if (ne > EDGE_CH) ne = EDGE_CH;
                gemm(EF + (size_t)r0 * 128, 128, 128, bw[b].pe0, P(preE[b] + 1), T1, ne, 256, 1);
                gemm(T1, 256, 256, bw[b].pe1, P(preE[b] + 3), EFPb + (size_t)r0 * 128, ne, 128, 1);
            }
            vml(U, P(preA[b]), P(preA[b] + 1), AT1, 128, 256, 1);
            vml(AT1, P(preA[b] + 2), P(preA[b] + 3), UPb, 256, 128, 1);
            EFp = EFPb; NFp = NFPb; Uc = UPb;
        }

        // fold u-contribution into the first-layer biases
        vml(Uc, P(convE[b]) + (size_t)384 * 256, P(convE[b] + 1), UB_E, 128, 256, 0);
        vml(Uc, P(convN[b]) + (size_t)256 * 256, P(convN[b] + 1), UB_N, 128, 256, 0);

        // edge MLP, concat layer decomposed:  Hsrc, Hdst once per node
        gemm(NFp, 128, 128, bw[b].esrc, nullptr, NS1, N_NODES, 256, 0);
        gemm(NFp, 128, 128, bw[b].edst, nullptr, NS2, N_NODES, 256, 0);

        fill(VE, (size_t)N_NODES * 128, 0.0f);
        fill(ESUM, 128, 0.0f);
        fill(NSUM, 128, 0.0f);

        for (int r0 = 0; r0 < N_EDGES; r0 += EDGE_CH) {
            int ne = N_EDGES - r0; if (ne > EDGE_CH) ne = EDGE_CH;
            gemm(EFp + (size_t)r0 * 128, 128, 128, bw[b].eef, UB_E, T1, ne, 256, 0);
            edge_gather_silu<<<(unsigned)(((size_t)ne * 64 + 255) / 256), 256, 0, stream>>>(
                T1, NS1, NS2, src, dst, r0, ne);
            gemm(T1, 256, 256, bw[b].e2, P(convE[b] + 3), T2, ne, 256, 1);
            gemm(T2, 256, 256, bw[b].e3, P(convE[b] + 5), EF2 + (size_t)r0 * 128, ne, 128, 1);
            scatter_edges<<<(unsigned)(((size_t)ne * 32 + 255) / 256), 256, 0, stream>>>(
                EF2 + (size_t)r0 * 128, dst, r0, ne, VE);
        }
        colsum_kernel<<<(N_EDGES + 511) / 512, 256, 0, stream>>>(EF2, N_EDGES, ESUM);
        ve_div<<<(unsigned)(((size_t)N_NODES * 128 + 255) / 256), 256, 0, stream>>>(VE, DEG, N_NODES);

        // node MLP, concat layer decomposed
        gemm(VE,  128, 128, bw[b].nve, UB_N, NT1, N_NODES, 256, 0);
        gemm(NFp, 128, 128, bw[b].nnf, nullptr, NT2, N_NODES, 256, 0);
        combine_silu<<<(unsigned)(((size_t)N_NODES * 256 + 255) / 256), 256, 0, stream>>>(
            NT1, NT2, (size_t)N_NODES * 256);
        gemm(NT1, 256, 256, bw[b].n2, P(convN[b] + 3), NT2, N_NODES, 256, 1);
        gemm(NT2, 256, 256, bw[b].n3, P(convN[b] + 5), NF2, N_NODES, 128, 1);
        colsum_kernel<<<(N_NODES + 511) / 512, 256, 0, stream>>>(NF2, N_NODES, NSUM);

        // attr MLP (M=1)
        assemble_avec<<<1, 384, 0, stream>>>(ESUM, NSUM, Uc, AVEC,
                                             1.0f / N_EDGES, 1.0f / N_NODES);
        vml(AVEC, P(convA[b]), P(convA[b] + 1), AT1, 384, 256, 1);
        vml(AT1, P(convA[b] + 2), P(convA[b] + 3), AT2, 256, 256, 1);
        vml(AT2, P(convA[b] + 4), P(convA[b] + 5), U2, 256, 128, 1);

        // skip connections
        axpy(EF, EF2, (size_t)N_EDGES * 128);
        axpy(NF, NF2, (size_t)N_NODES * 128);
        axpy(U, U2, 128);
    }

    // ---------------- set2set readouts ------------------------------------
    auto set2set = [&](const float* feat, int n, int pb, float* qst) {
        const float* Whh = P(pb);
        const float* Wih = P(pb + 1);
        const float* bhh = P(pb + 2);
        const float* bih = P(pb + 3);
        fill(Hs, 128, 0.0f);
        fill(Cs, 128, 0.0f);
        fill(qst, 256, 0.0f);
        for (int it = 0; it < 3; ++it) {
            lstm_step<<<1, 512, 0, stream>>>(qst, Wih, Whh, bih, bhh, Hs, Cs);
            fill(GMAX, 1, -3.0e38f);
            s2s_logits<<<(n + 255) / 256, 256, 0, stream>>>(feat, n, Hs, LOGI, GMAX);
            fill(RACC, 128, 0.0f);
            fill(PSUM, 1, 0.0f);
            s2s_accum<<<(n + 255) / 256, 256, 0, stream>>>(feat, LOGI, n, GMAX, RACC, PSUM);
            qstar_update<<<1, 256, 0, stream>>>(Hs, RACC, PSUM, qst);
        }
    };
    set2set(NF, N_NODES, NODE_S2S, NODEQ);
    set2set(EF, N_EDGES, EDGE_S2S, EDGEQ);

    // ---------------- final regression head -------------------------------
    out_mlp<<<1, 128, 0, stream>>>(NODEQ, EDGEQ, U,
                                   P(OUT_MLP), P(OUT_MLP + 1),
                                   P(OUT_MLP + 2), P(OUT_MLP + 3),
                                   P(OUT_MLP + 4), P(OUT_MLP + 5),
                                   (float*)d_out);
}
